// SDEFunc_52003464020797
// MI455X (gfx1250) — compile-verified, matrix operand order checked by reading, not run on hardware
//
#include <hip/hip_runtime.h>
#include <hip/hip_bf16.h>
#include <stdint.h>

// ---------------------------------------------------------------------------
// Types
// ---------------------------------------------------------------------------
typedef __bf16 bf16_t;
typedef __attribute__((ext_vector_type(16))) __bf16   v16bf;
typedef __attribute__((ext_vector_type(8)))  float    v8f;
typedef __attribute__((ext_vector_type(4))) uint32_t v4u;
typedef __attribute__((ext_vector_type(8)))  int      v8i;
typedef __attribute__((ext_vector_type(4)))  int      v4i;

#define S_    128
#define B_    512
#define H_    128
#define STEPS 127
#define MT    16      // batch rows per workgroup
#define NT    256     // threads per workgroup = 8 waves (wave32)

#if defined(__has_builtin)
#if __has_builtin(__builtin_amdgcn_tensor_load_to_lds) && __has_builtin(__builtin_amdgcn_s_wait_tensorcnt)
#define HAS_TDM 1
#endif
#endif

// ---------------------------------------------------------------------------
// LDS layout (bytes).  Total ~305 KB < 320 KB WGP LDS.
// ---------------------------------------------------------------------------
constexpr int OFF_WIH  = 0;                              // 384x128 bf16
constexpr int OFF_WHH  = OFF_WIH + 384 * 128 * 2;        // 384x128 bf16
constexpr int OFF_DW1  = OFF_WHH + 384 * 128 * 2;        // 64x128 bf16
constexpr int OFF_GW1  = OFF_DW1 + 64 * 128 * 2;
constexpr int OFF_DW2  = OFF_GW1 + 64 * 128 * 2;         // 128x64 bf16
constexpr int OFF_GW2  = OFF_DW2 + 128 * 64 * 2;
constexpr int OFF_XST  = OFF_GW2 + 128 * 64 * 2;         // 16x128 f32 (TDM stage)
constexpr int OFF_H2   = OFF_XST + MT * H_ * 4;          // 16x128 f32 carry h2
constexpr int OFF_H1   = OFF_H2 + MT * H_ * 4;           // carry h1
constexpr int OFF_HG   = OFF_H1 + MT * H_ * 4;           // post-GRU h
constexpr int OFF_HS   = OFF_HG + MT * H_ * 4;           // SDE working h
constexpr int OFF_T1   = OFF_HS + MT * H_ * 4;           // 16x64 bf16 (tanh mid)
constexpr int OFF_U1   = OFF_T1 + MT * 64 * 2;           // 16x64 bf16 (softplus mid)
constexpr int OFF_BIH  = OFF_U1 + MT * 64 * 2;           // 384 f32
constexpr int OFF_BHH  = OFF_BIH + 384 * 4;
constexpr int OFF_DB1  = OFF_BHH + 384 * 4;              // 64 f32
constexpr int OFF_GB1  = OFF_DB1 + 64 * 4;
constexpr int OFF_DB2  = OFF_GB1 + 64 * 4;               // 128 f32
constexpr int OFF_GB2  = OFF_DB2 + 128 * 4;
constexpr int OFF_MISC = OFF_GB2 + 128 * 4;              // 5 x 16 ints/floats
constexpr int SMEM_BYTES = OFF_MISC + 16 * 4 * 5;

// ---------------------------------------------------------------------------
// Helpers
// ---------------------------------------------------------------------------
__device__ __forceinline__ v8f wmma_bf(v16bf a, v16bf b, v8f c) {
  // v_wmma_f32_16x16x32_bf16
  return __builtin_amdgcn_wmma_f32_16x16x32_bf16(false, a, false, b, (short)0, c,
                                                 false, false);
}

// K offset inside a 32-wide K block for dword d of a 16-bit A/B fragment.
// ISA 7.12.2: lane group 0 holds K 0-7 (v0-3) and 16-23 (v4-7); group 1 holds
// K 8-15 and 24-31.  Each dword packs {K, K+1}.
__device__ __forceinline__ int frag_k(int d, int grp) {
  return (d < 4) ? (grp * 8 + 2 * d) : (16 + grp * 8 + 2 * (d - 4));
}

// Fragment from a bf16 row-major [rows][ldk] LDS tile; row = lane & 15
// (callers pre-offset the base pointer for B tiles: base + n0*ldk).
__device__ __forceinline__ v16bf frag16_bf(const bf16_t* base, int ldk, int k0) {
  const int lane = threadIdx.x & 31;
  const int grp  = lane >> 4;
  const bf16_t* r = base + (lane & 15) * ldk + k0;
  v16bf a;
#pragma unroll
  for (int d = 0; d < 8; ++d) {
    int k = frag_k(d, grp);
    a[2 * d]     = r[k];
    a[2 * d + 1] = r[k + 1];
  }
  return a;
}

// Same, but source is f32 in LDS; convert to bf16 on the fly.
__device__ __forceinline__ v16bf frag16_f32(const float* base, int ldk, int k0) {
  const int lane = threadIdx.x & 31;
  const int grp  = lane >> 4;
  const float* r = base + (lane & 15) * ldk + k0;
  v16bf a;
#pragma unroll
  for (int d = 0; d < 8; ++d) {
    int k = frag_k(d, grp);
    a[2 * d]     = (bf16_t)r[k];
    a[2 * d + 1] = (bf16_t)r[k + 1];
  }
  return a;
}

__device__ __forceinline__ float sigm(float x) { return 1.f / (1.f + __expf(-x)); }
__device__ __forceinline__ float splus(float x) {
  return (x > 20.f) ? x : log1pf(__expf(x));
}

// Deterministic per-(counter,row) standard normal (splitmix64 + Box-Muller).
__device__ __forceinline__ float nrand(uint32_t ctr, uint32_t idx) {
  uint64_t x = (((uint64_t)ctr << 32) | idx) ^ 0x9E3779B97F4A7C15ull;
  x ^= x >> 30; x *= 0xBF58476D1CE4E5B9ull;
  x ^= x >> 27; x *= 0x94D049BB133111EBull;
  x ^= x >> 31;
  uint32_t a = (uint32_t)x, b = (uint32_t)(x >> 32);
  float u1 = (a + 0.5f) * 2.3283064365386963e-10f;  // (0,1)
  float u2 = b * 2.3283064365386963e-10f;           // [0,1)
  float r  = sqrtf(-2.f * __logf(u1));
  return r * __cosf(6.283185307179586f * u2);
}

// ---------------------------------------------------------------------------
// Prologue: per-step time statistics td[i] (reduction over B), into d_ws.
// ---------------------------------------------------------------------------
__global__ void td_prep_kernel(const float* __restrict__ times,
                               const int* __restrict__ nz,
                               float* __restrict__ td_out) {
  const int i   = blockIdx.x;     // step 0..STEPS-1
  const int tid = threadIdx.x;    // 256 threads
  __shared__ float s_ts[256], s_te[256];
  __shared__ int   s_c[256];
  float ts = 0.f, te = 0.f;
  int   c  = 0;
  for (int b = tid; b < B_; b += 256) {
    bool  mask  = i < nz[b] - 1;
    float tnext = times[b * S_ + i + 1];
    bool  valid = mask && (tnext != 0.f);
    if (valid) { ts += times[b * S_ + i]; te += tnext; c++; }
  }
  s_ts[tid] = ts; s_te[tid] = te; s_c[tid] = c;
  __syncthreads();
  for (int o = 128; o > 0; o >>= 1) {
    if (tid < o) {
      s_ts[tid] += s_ts[tid + o];
      s_te[tid] += s_te[tid + o];
      s_c[tid]  += s_c[tid + o];
    }
    __syncthreads();
  }
  if (tid == 0) {
    float cnt = (float)(s_c[0] < 1 ? 1 : s_c[0]);
    float tsv = s_ts[0] / cnt * 0.02f;
    float tev = s_te[0] / cnt * 0.02f;
    td_out[i] = (tev == tsv) ? 1.0f : (tev - tsv);
  }
}

// ---------------------------------------------------------------------------
// Main: each workgroup owns 16 batch rows and runs the full 127-step scan
// with all weights (bf16) resident in LDS.  No grid sync needed.
// ---------------------------------------------------------------------------
__global__ void __launch_bounds__(NT, 1)
sde_scan_kernel(const float* __restrict__ inputs, const float* __restrict__ times,
                const int* __restrict__ nz,
                const float* __restrict__ Wih, const float* __restrict__ Whh,
                const float* __restrict__ bih, const float* __restrict__ bhh,
                const float* __restrict__ dW1, const float* __restrict__ db1,
                const float* __restrict__ dW2, const float* __restrict__ db2,
                const float* __restrict__ gW1, const float* __restrict__ gb1,
                const float* __restrict__ gW2, const float* __restrict__ gb2,
                float* __restrict__ out, const float* __restrict__ td_arr) {
  extern __shared__ __align__(16) char smem[];
  const int tid  = threadIdx.x;
  const int b0   = blockIdx.x * MT;
  const int w    = tid >> 5;          // wave index 0..7
  const int lane = tid & 31;
  const int col  = lane & 15;         // C-tile column (ISA C/D layout)
  const int mo   = (lane >> 4) << 3;  // +8 row offset for lanes 16..31

  bf16_t* WihB = (bf16_t*)(smem + OFF_WIH);
  bf16_t* WhhB = (bf16_t*)(smem + OFF_WHH);
  bf16_t* dW1B = (bf16_t*)(smem + OFF_DW1);
  bf16_t* gW1B = (bf16_t*)(smem + OFF_GW1);
  bf16_t* dW2B = (bf16_t*)(smem + OFF_DW2);
  bf16_t* gW2B = (bf16_t*)(smem + OFF_GW2);
  float*  xst  = (float*)(smem + OFF_XST);
  float*  h2   = (float*)(smem + OFF_H2);
  float*  h1   = (float*)(smem + OFF_H1);
  float*  hg   = (float*)(smem + OFF_HG);
  float*  hs   = (float*)(smem + OFF_HS);
  bf16_t* t1B  = (bf16_t*)(smem + OFF_T1);
  bf16_t* u1B  = (bf16_t*)(smem + OFF_U1);
  float*  bihL = (float*)(smem + OFF_BIH);
  float*  bhhL = (float*)(smem + OFF_BHH);
  float*  db1L = (float*)(smem + OFF_DB1);
  float*  gb1L = (float*)(smem + OFF_GB1);
  float*  db2L = (float*)(smem + OFF_DB2);
  float*  gb2L = (float*)(smem + OFF_GB2);
  int*    maskrow  = (int*)(smem + OFF_MISC);
  int*    validrow = maskrow + 16;
  int*    idxrow   = validrow + 16;
  int*    nzrow    = idxrow + 16;
  float*  zrow     = (float*)(nzrow + 16);

  // ---- one-time preamble: weights f32 -> bf16 LDS, biases, zero carries ----
  for (int e = tid; e < 384 * 128; e += NT) WihB[e] = (bf16_t)Wih[e];
  for (int e = tid; e < 384 * 128; e += NT) WhhB[e] = (bf16_t)Whh[e];
  for (int e = tid; e < 64 * 128; e += NT) { dW1B[e] = (bf16_t)dW1[e]; gW1B[e] = (bf16_t)gW1[e]; }
  for (int e = tid; e < 128 * 64; e += NT) { dW2B[e] = (bf16_t)dW2[e]; gW2B[e] = (bf16_t)gW2[e]; }
  for (int e = tid; e < 384; e += NT) { bihL[e] = bih[e]; bhhL[e] = bhh[e]; }
  for (int e = tid; e < 64;  e += NT) { db1L[e] = db1[e]; gb1L[e] = gb1[e]; }
  for (int e = tid; e < 128; e += NT) { db2L[e] = db2[e]; gb2L[e] = gb2[e]; }
  for (int e = tid; e < MT * H_; e += NT) { h2[e] = 0.f; h1[e] = 0.f; }
  if (tid < MT) {
    int v  = nz[b0 + tid];
    int id = v - 1;
    if (id < 0) id = 0;
    if (id > S_ - 1) id = S_ - 1;
    idxrow[tid] = id;
    nzrow[tid]  = v;
  }
  __syncthreads();

  float td_prev = 0.f;
  for (int i = 0; i < STEPS; ++i) {
    // ---- phase A: per-row flags + DMA x tile into LDS --------------------
    if (tid < MT) {
      bool  mk = i < nzrow[tid] - 1;
      float tn = times[(b0 + tid) * S_ + i + 1];
      maskrow[tid]  = mk ? 1 : 0;
      validrow[tid] = (mk && tn != 0.f) ? 1 : 0;
    }
#ifdef HAS_TDM
    if (w == 0) {
      // Tensor DMA: 2D tile 128 x 16 f32 (contiguous rows of inputs[i])
      uint64_t gaddr = (uint64_t)(uintptr_t)(inputs + ((size_t)i * B_ + b0) * H_);
      v4u g0;
      g0.x = 1u;                                   // count=1, user descriptor
      g0.y = (uint32_t)OFF_XST;                    // lds_addr
      g0.z = (uint32_t)(gaddr & 0xFFFFFFFFu);
      g0.w = (uint32_t)((gaddr >> 32) & 0x01FFFFFFu) | (2u << 30);  // type=2
      v8i g1;
      g1[0] = (int)(2u << 16);                                      // data_size=4B
      g1[1] = (int)(((uint32_t)H_ & 0xFFFFu) << 16);                // tensor_dim0 lo
      g1[2] = (int)(((uint32_t)H_ >> 16) | ((65536u & 0xFFFFu) << 16)); // dim0 hi | dim1 lo
      g1[3] = (int)((65536u >> 16) | ((uint32_t)H_ << 16));         // dim1 hi | tile_dim0
      g1[4] = (int)((uint32_t)MT);                                  // tile_dim1=16
      g1[5] = (int)((uint32_t)H_);                                  // dim0_stride lo32
      g1[6] = 0;
      g1[7] = 0;
      v4i g2 = {0, 0, 0, 0}, g3 = {0, 0, 0, 0};
      v8i g4 = {0, 0, 0, 0, 0, 0, 0, 0};   // 6-arg toolchain: extra group
      __builtin_amdgcn_tensor_load_to_lds(g0, g1, g2, g3, g4, 0);
      __builtin_amdgcn_s_wait_tensorcnt((short)0);
    }
#else
    for (int e = tid; e < MT * H_; e += NT)
      xst[e] = inputs[((size_t)i * B_ + b0) * H_ + e];
#endif
    float td = td_arr[i];
    float dt = td * 0.1f;
    float sq = sqrtf(dt);
    __syncthreads();

    // ---- GRU: wave w owns 16-col slice cb of H for all three gates -------
    {
      const int cb = w * 16;
      v8f air = {}, aiz = {}, ain = {}, ahr = {}, ahz = {}, ahn = {};
#pragma unroll
      for (int kb = 0; kb < 4; ++kb) {
        const int k0 = kb * 32;
        v16bf ax = frag16_f32(xst, H_, k0);
        v16bf ah = frag16_f32(h2, H_, k0);
        air = wmma_bf(ax, frag16_bf(WihB + (cb)*H_,        H_, k0), air);
        aiz = wmma_bf(ax, frag16_bf(WihB + (128 + cb)*H_,  H_, k0), aiz);
        ain = wmma_bf(ax, frag16_bf(WihB + (256 + cb)*H_,  H_, k0), ain);
        ahr = wmma_bf(ah, frag16_bf(WhhB + (cb)*H_,        H_, k0), ahr);
        ahz = wmma_bf(ah, frag16_bf(WhhB + (128 + cb)*H_,  H_, k0), ahz);
        ahn = wmma_bf(ah, frag16_bf(WhhB + (256 + cb)*H_,  H_, k0), ahn);
      }
#pragma unroll
      for (int v = 0; v < 8; ++v) {
        const int m = mo + v;
        const int c = cb + col;
        float rr = sigm(air[v] + bihL[c] + ahr[v] + bhhL[c]);
        float zz = sigm(aiz[v] + bihL[128 + c] + ahz[v] + bhhL[128 + c]);
        float nn = tanhf(ain[v] + bihL[256 + c] + rr * (ahn[v] + bhhL[256 + c]));
        float hold = h2[m * H_ + c];
        float hnew = (1.f - zz) * nn + zz * hold;
        float hv   = maskrow[m] ? hnew : hold;
        hg[m * H_ + c] = hv;
        hs[m * H_ + c] = hv;
      }
    }
    __syncthreads();

    // ---- 10 Euler-Maruyama substeps --------------------------------------
    for (int s5 = 0; s5 < 10; ++s5) {
      if (tid < MT) zrow[tid] = nrand((uint32_t)(i * 10 + s5), (uint32_t)(b0 + tid));
      {
        // MLP layer 1: waves 0-3 drift(tanh), waves 4-7 diffusion(softplus)
        const int half = w >> 2;
        const int n0   = (w & 3) * 16;
        const bf16_t* W1 = half ? gW1B : dW1B;
        const float*  bb = half ? gb1L : db1L;
        bf16_t*       ot = half ? u1B : t1B;
        v8f acc = {};
#pragma unroll
        for (int kb = 0; kb < 4; ++kb)
          acc = wmma_bf(frag16_f32(hs, H_, kb * 32),
                        frag16_bf(W1 + n0 * H_, H_, kb * 32), acc);
#pragma unroll
        for (int v = 0; v < 8; ++v) {
          const int m = mo + v;
          float x = acc[v] + bb[n0 + col];
          float y = half ? splus(x) : tanhf(x);
          ot[m * 64 + n0 + col] = (bf16_t)y;
        }
      }
      __syncthreads();
      {
        // MLP layer 2: wave w owns 16-col slice of H, drift + diffusion
        const int n0 = w * 16;
        v8f af = {}, ag = {};
#pragma unroll
        for (int kb = 0; kb < 2; ++kb) {
          af = wmma_bf(frag16_bf(t1B, 64, kb * 32),
                       frag16_bf(dW2B + n0 * 64, 64, kb * 32), af);
          ag = wmma_bf(frag16_bf(u1B, 64, kb * 32),
                       frag16_bf(gW2B + n0 * 64, 64, kb * 32), ag);
        }
#pragma unroll
        for (int v = 0; v < 8; ++v) {
          const int m = mo + v;
          const int c = n0 + col;
          float f = af[v] + db2L[c];
          float g = ag[v] + gb2L[c];
          hs[m * H_ + c] += f * dt + g * (sq * zrow[m]);
        }
      }
      __syncthreads();
    }

    // ---- finalize step: update carries, emit output rows -----------------
    {
      const float u = __expf(-td_prev);
#pragma unroll
      for (int j = 0; j < 8; ++j) {
        const int e = tid * 8 + j;      // 2048 elements
        const int m = e >> 7;
        const int c = e & (H_ - 1);
        float hgv = hg[e], hsv = hs[e], h1p = h1[e];
        bool  vd  = validrow[m] != 0;
        float h1n = vd ? hsv : 0.f;
        float h2n = (i == 0) ? hgv : (vd ? (u * hgv + (1.f - u) * h1p) : 0.f);
        h1[e] = h1n;
        h2[e] = h2n;
        if (i + 1 == idxrow[m]) out[(size_t)(b0 + m) * H_ + c] = h2n;
      }
    }
    td_prev = td;
    __syncthreads();
  }
}

// ---------------------------------------------------------------------------
// Launch
// ---------------------------------------------------------------------------
extern "C" void kernel_launch(void* const* d_in, const int* in_sizes, int n_in,
                              void* d_out, int out_size, void* d_ws, size_t ws_size,
                              hipStream_t stream) {
  const float* inputs = (const float*)d_in[0];
  const float* times  = (const float*)d_in[1];
  const int*   nzp    = (const int*)d_in[2];
  const float* Wih = (const float*)d_in[3];
  const float* Whh = (const float*)d_in[4];
  const float* bih = (const float*)d_in[5];
  const float* bhh = (const float*)d_in[6];
  const float* dW1 = (const float*)d_in[7];
  const float* db1 = (const float*)d_in[8];
  const float* dW2 = (const float*)d_in[9];
  const float* db2 = (const float*)d_in[10];
  const float* gW1 = (const float*)d_in[11];
  const float* gb1 = (const float*)d_in[12];
  const float* gW2 = (const float*)d_in[13];
  const float* gb2 = (const float*)d_in[14];
  float* out = (float*)d_out;
  float* td  = (float*)d_ws;   // STEPS floats of scratch

  td_prep_kernel<<<STEPS, 256, 0, stream>>>(times, nzp, td);
  sde_scan_kernel<<<B_ / MT, NT, SMEM_BYTES, stream>>>(
      inputs, times, nzp, Wih, Whh, bih, bhh, dW1, db1, dW2, db2, gW1, gb1, gW2,
      gb2, out, td);
}
